// Attention_27487790694937
// MI455X (gfx1250) — matrix operand order repaired and checked by reading, not compiled, not run on hardware
//
#include <hip/hip_runtime.h>
#include <hip/hip_bf16.h>

// ---------------------------------------------------------------------------
// Distance-attention for MI455X (gfx1250), wave32 + v_wmma_f32_16x16x32_bf16.
//   q,k,v = x @ W{q,k,v}^T        (bf16 WMMA, f32 accum)
//   att   = softmax(cdist(q,k)/sqrt(D))  -> flash-style, never hits HBM
//   out   = att @ v
// ---------------------------------------------------------------------------

static constexpr int Bz  = 16;
static constexpr int Nn  = 2048;
static constexpr int Dd  = 512;
static constexpr int BN  = Bz * Nn;          // 32768 rows total
static constexpr int KB  = 128;              // keys per streamed block

typedef __bf16 bf16;
typedef __attribute__((ext_vector_type(16))) __bf16 v16bf;
typedef __attribute__((ext_vector_type(8)))  __bf16 v8bf;
typedef __attribute__((ext_vector_type(8)))  float  v8f;
typedef __attribute__((ext_vector_type(4)))  int    v4i;

// ---- CDNA5 async global->LDS path (guarded; falls back to plain copies) ----
#if __has_builtin(__builtin_amdgcn_global_load_async_to_lds_b128)
#define USE_ASYNC_LDS 1
#else
#define USE_ASYNC_LDS 0
#endif

__device__ __forceinline__ void wait_async0() {
#if __has_builtin(__builtin_amdgcn_s_wait_asynccnt)
  __builtin_amdgcn_s_wait_asynccnt(0);
#else
  asm volatile("s_wait_asynccnt 0" ::: "memory");
#endif
}

__device__ __forceinline__ v8f wmma_bf16(v16bf a, v16bf b, v8f c) {
  // 8 args: (neg_a, A, neg_b, B, c_mod, C, reuse_a, reuse_b)
  return __builtin_amdgcn_wmma_f32_16x16x32_bf16(false, a, false, b, (short)0, c, false, false);
}

// A-matrix 16x32 bf16 fragment from row-major LDS (rows 0..15, ld elements/row).
// Lane l<16: M=l, K = k0+{0..15}; lane l>=16: M=l-16, K = k0+{16..31}.
__device__ __forceinline__ v16bf a_frag_lds(const bf16* S, int ld, int k0) {
  int lane = threadIdx.x & 31;
  int m    = lane & 15;
  int sel  = (lane & 16) ? 16 : 0;
  const bf16* p = S + m * ld + k0 + sel;
  v8bf lo = *(const v8bf*)(p);        // K = k0+sel .. +7     (16B aligned)
  v8bf hi = *(const v8bf*)(p + 8);    // K = k0+sel+8 .. +7
  return __builtin_shufflevector(lo, hi, 0,1,2,3,4,5,6,7,8,9,10,11,12,13,14,15);
}

// B-matrix 32x16 bf16 fragment where column n of B is row (n0+n) of S (row-major,
// ld elements/row), i.e. B[k][n] = S[n0+n][k0+k].  Lane half selects K 0-15 / 16-31.
__device__ __forceinline__ v16bf b_frag_lds(const bf16* S, int ld, int n0, int k0) {
  int lane = threadIdx.x & 31;
  int n    = lane & 15;
  int sel  = (lane & 16) ? 16 : 0;
  const bf16* p = S + (n0 + n) * ld + k0 + sel;
  v8bf lo = *(const v8bf*)(p);
  v8bf hi = *(const v8bf*)(p + 8);
  return __builtin_shufflevector(lo, hi, 0,1,2,3,4,5,6,7,8,9,10,11,12,13,14,15);
}

// Same B fragment from pre-converted bf16 global weights (row-major [out][in]).
__device__ __forceinline__ v16bf b_frag_bf16_global(const bf16* W, int ld, int n0, int k0) {
  int lane = threadIdx.x & 31;
  int n    = lane & 15;
  int sel  = (lane & 16) ? 16 : 0;
  const bf16* p = W + (size_t)(n0 + n) * ld + k0 + sel;
  v8bf lo = *(const v8bf*)(p);
  v8bf hi = *(const v8bf*)(p + 8);
  return __builtin_shufflevector(lo, hi, 0,1,2,3,4,5,6,7,8,9,10,11,12,13,14,15);
}

// ---------------------------------------------------------------------------
// Kernel 0: one-time fp32 -> bf16 conversion of the three weight matrices.
// grid = (Dd*Dd/8)/256 = 128 blocks.
// ---------------------------------------------------------------------------
__global__ __launch_bounds__(256)
void wcvt_kernel(const float* __restrict__ Wq, const float* __restrict__ Wk,
                 const float* __restrict__ Wv,
                 bf16* __restrict__ Wqb, bf16* __restrict__ Wkb, bf16* __restrict__ Wvb) {
  const int idx = blockIdx.x * 256 + threadIdx.x;     // one v8bf per thread
  const float* srcs[3] = {Wq, Wk, Wv};
  bf16*        dsts[3] = {Wqb, Wkb, Wvb};
#pragma unroll
  for (int m = 0; m < 3; ++m) {
    const float4* s = (const float4*)srcs[m] + (size_t)idx * 2;
    float4 a = s[0], c = s[1];
    v8bf d;
    d[0] = (bf16)a.x; d[1] = (bf16)a.y; d[2] = (bf16)a.z; d[3] = (bf16)a.w;
    d[4] = (bf16)c.x; d[5] = (bf16)c.y; d[6] = (bf16)c.z; d[7] = (bf16)c.w;
    *(v8bf*)(dsts[m] + (size_t)idx * 8) = d;
  }
}

// ---------------------------------------------------------------------------
// Kernel 1: QKV projection.  grid = BN/16 blocks, 256 thr (8 wave32).
// x tile converted to bf16 once at staging; weight fragments are raw b128
// loads of pre-converted bf16 (L2-resident), feeding WMMA directly.
// Q,K stored row-major bf16; V stored TRANSPOSED [B][D][N] bf16 so the
// attention kernel's V stage is a contiguous copy.
// ---------------------------------------------------------------------------
__global__ __launch_bounds__(256)
void qkv_kernel(const float* __restrict__ x,
                const bf16* __restrict__ Wqb, const bf16* __restrict__ Wkb,
                const bf16* __restrict__ Wvb,
                bf16* __restrict__ Qb, bf16* __restrict__ Kb, bf16* __restrict__ Vtg) {
  __shared__ bf16 xs[16 * Dd];                        // 16 KB
  const int tid  = threadIdx.x;
  const int row0 = blockIdx.x * 16;

  {
    const float4* src = (const float4*)(x + (size_t)row0 * Dd);
    for (int i = tid; i < 16 * Dd / 4; i += 256) {
      float4 v = src[i];
      bf16* d = xs + i * 4;
      d[0] = (bf16)v.x; d[1] = (bf16)v.y; d[2] = (bf16)v.z; d[3] = (bf16)v.w;
    }
  }
  __syncthreads();

  const int w    = tid >> 5;
  const int lane = tid & 31;
  const int n    = lane & 15;
  const int mb   = (lane & 16) ? 8 : 0;
  const int b    = row0 / Nn;
  const int r0   = row0 - b * Nn;                     // row within batch

  for (int ct = 0; ct < 4; ++ct) {
    const int n0 = (w * 4 + ct) * 16;                 // output-feature tile base
    v8f aq = {}, ak = {}, av = {};
#pragma unroll 4
    for (int k0 = 0; k0 < Dd; k0 += 32) {
      v16bf a = a_frag_lds(xs, Dd, k0);
      aq = wmma_bf16(a, b_frag_bf16_global(Wqb, Dd, n0, k0), aq);
      ak = wmma_bf16(a, b_frag_bf16_global(Wkb, Dd, n0, k0), ak);
      av = wmma_bf16(a, b_frag_bf16_global(Wvb, Dd, n0, k0), av);
    }
    const size_t base = (size_t)(row0 + mb) * Dd + n0 + n;
#pragma unroll
    for (int j = 0; j < 8; ++j) {
      Qb[base + (size_t)j * Dd] = (bf16)aq[j];
      Kb[base + (size_t)j * Dd] = (bf16)ak[j];
    }
    v8bf vv;                                          // V transposed: contiguous rows
#pragma unroll
    for (int j = 0; j < 8; ++j) vv[j] = (bf16)av[j];
    *(v8bf*)(Vtg + ((size_t)b * Dd + n0 + n) * Nn + r0 + mb) = vv;
  }
}

// ---------------------------------------------------------------------------
// Kernel 2: per-row squared norms of Q and K (wave per row).
// ---------------------------------------------------------------------------
__global__ __launch_bounds__(256)
void norms_kernel(const bf16* __restrict__ Qb, const bf16* __restrict__ Kb,
                  float* __restrict__ q2, float* __restrict__ k2) {
  const int row  = blockIdx.x * 8 + (threadIdx.x >> 5);
  const int lane = threadIdx.x & 31;
  const bf16* qp = Qb + (size_t)row * Dd;
  const bf16* kp = Kb + (size_t)row * Dd;
  float sq = 0.f, sk = 0.f;
  for (int i = lane; i < Dd; i += 32) {
    float a = (float)qp[i]; sq += a * a;
    float c = (float)kp[i]; sk += c * c;
  }
#pragma unroll
  for (int m = 16; m >= 1; m >>= 1) {
    sq += __shfl_xor(sq, m, 32);
    sk += __shfl_xor(sk, m, 32);
  }
  if (lane == 0) { q2[row] = sq; k2[row] = sk; }
}

// ---------------------------------------------------------------------------
// Kernel 3: fused distance-softmax attention (flash style; exp argument is
// bounded by ~2 so no running max is needed).  grid = BN/16 blocks, 256 thr
// (8 wave32), ~283 KB dynamic LDS (1 WG per 320 KB WGP -> big VGPR budget,
// so all 16 Q A-fragments are held in registers for the whole kernel).
// ---------------------------------------------------------------------------
__global__ __launch_bounds__(256)
void attn_kernel(const bf16* __restrict__ Qb, const bf16* __restrict__ Kb,
                 const bf16* __restrict__ Vtg,
                 const float* __restrict__ q2, const float* __restrict__ k2,
                 float* __restrict__ out) {
  extern __shared__ char smem[];
  bf16*  Qs  = (bf16*)smem;                 // [16][Dd]
  bf16*  Ks  = Qs + 16 * Dd;                // [KB][Dd]   key-major
  bf16*  Vt  = Ks + KB * Dd;                // [Dd][KB]   d-major
  bf16*  Ps  = Vt + (size_t)Dd * KB;        // [16][KB]   probabilities (bf16)
  float* rs  = (float*)(Ps + 16 * KB);      // [16] running row sums
  float* q2s = rs + 16;                     // [16]

  const int   tid   = threadIdx.x;
  const int   w     = tid >> 5;
  const int   lane  = tid & 31;
  const int   n     = lane & 15;
  const int   mb    = (lane & 16) ? 8 : 0;
  const int   grow0 = blockIdx.x * 16;      // global query-row tile
  const int   b     = grow0 / Nn;           // batch (tiles never straddle batches)
  const float inv_scale = 0.04419417382415922f;   // 1/sqrt(512)

  // Stage Q tile + row state, then hoist all Q A-fragments into registers.
  {
    const v8bf* qsrc = (const v8bf*)(Qb + (size_t)grow0 * Dd);
    v8bf* qdst = (v8bf*)Qs;
    for (int i = tid; i < 16 * Dd / 8; i += 256) qdst[i] = qsrc[i];
    if (tid < 16) { rs[tid] = 0.f; q2s[tid] = q2[grow0 + tid]; }
  }
  __syncthreads();
  v16bf qf[16];
#pragma unroll
  for (int kk = 0; kk < 16; ++kk) qf[kk] = a_frag_lds(Qs, Dd, kk * 32);

  v8f o[4] = {v8f{}, v8f{}, v8f{}, v8f{}};  // 16x64 output slice per wave

  const bf16* vbat = Vtg + (size_t)b * Dd * Nn;   // [Dd][Nn] for this batch

  for (int kb = 0; kb < Nn / KB; ++kb) {
    __syncthreads();                        // prev PV done reading Ks/Vt
    const int keybase = b * Nn + kb * KB;
    const bf16* ksrc = Kb + (size_t)keybase * Dd;
    const bf16* vsrc = vbat + kb * KB;      // row d starts at vsrc + d*Nn

#if USE_ASYNC_LDS
    for (int i = tid; i < KB * Dd / 8; i += 256)
      __builtin_amdgcn_global_load_async_to_lds_b128(
          (v4i*)(ksrc + (size_t)i * 8), (v4i*)(Ks + (size_t)i * 8), 0, 0);
    for (int i = tid; i < Dd * (KB / 8); i += 256) {
      const int d = i >> 4, c = i & 15;     // KB/8 == 16 chunks per row
      __builtin_amdgcn_global_load_async_to_lds_b128(
          (v4i*)(vsrc + (size_t)d * Nn + c * 8), (v4i*)(Vt + d * KB + c * 8), 0, 0);
    }
    wait_async0();
#else
    {
      const v8bf* src = (const v8bf*)ksrc;
      v8bf* dst = (v8bf*)Ks;
      for (int i = tid; i < KB * Dd / 8; i += 256) dst[i] = src[i];
    }
    for (int i = tid; i < Dd * (KB / 8); i += 256) {
      const int d = i >> 4, c = i & 15;
      *(v8bf*)(Vt + d * KB + c * 8) = *(const v8bf*)(vsrc + (size_t)d * Nn + c * 8);
    }
#endif
    __syncthreads();

    // --- S phase: wave w computes S tile for keys [w*16, w*16+16) -----------
    v8f s = {};
#pragma unroll
    for (int kk = 0; kk < 16; ++kk) {
      v16bf bb = b_frag_lds(Ks, Dd, w * 16, kk * 32);
      s = wmma_bf16(qf[kk], bb, s);
    }
    const float k2v = k2[keybase + w * 16 + n];
    float pv[8];
#pragma unroll
    for (int j = 0; j < 8; ++j) {
      float d2 = q2s[mb + j] + k2v - 2.0f * s[j];
      float dd = sqrtf(fmaxf(d2, 0.0f));
      pv[j] = __expf(dd * inv_scale);       // softmax numerator (arg <= ~2)
    }
#pragma unroll
    for (int j = 0; j < 8; ++j) {
      float t = pv[j];
#pragma unroll
      for (int m = 8; m >= 1; m >>= 1) t += __shfl_xor(t, m, 16);
      if (n == 0) atomicAdd(&rs[mb + j], t);          // ds_add_f32
      Ps[(mb + j) * KB + w * 16 + n] = (bf16)pv[j];
    }
    __syncthreads();

    // Prefetch next key block while the PV WMMAs run (global_prefetch_b8).
    if (kb + 1 < Nn / KB) {
      __builtin_prefetch(ksrc + (size_t)KB * Dd + (size_t)tid * 512, 0, 0);
      __builtin_prefetch(vsrc + KB + (size_t)(tid >> 1) * Nn, 0, 0);
    }

    // --- PV phase: wave w accumulates output cols [w*64, w*64+64) -----------
#pragma unroll
    for (int t = 0; t < 4; ++t) {
      const int c0 = w * 64 + t * 16;
#pragma unroll
      for (int k0 = 0; k0 < KB; k0 += 32) {
        v16bf a  = a_frag_lds(Ps, KB, k0);
        v16bf bb = b_frag_lds(Vt, KB, c0, k0);
        o[t] = wmma_bf16(a, bb, o[t]);
      }
    }
  }
  __syncthreads();                          // row-sum atomics visible

  // Normalize and store f32 output.
#pragma unroll
  for (int t = 0; t < 4; ++t) {
    const int c = w * 64 + t * 16 + n;
#pragma unroll
    for (int j = 0; j < 8; ++j) {
      out[(size_t)(grow0 + mb + j) * Dd + c] = o[t][j] / rs[mb + j];
    }
  }
}

// ---------------------------------------------------------------------------
extern "C" void kernel_launch(void* const* d_in, const int* in_sizes, int n_in,
                              void* d_out, int out_size, void* d_ws, size_t ws_size,
                              hipStream_t stream) {
  const float* x  = (const float*)d_in[0];
  const float* Wq = (const float*)d_in[1];
  const float* Wk = (const float*)d_in[2];
  const float* Wv = (const float*)d_in[3];
  float* out = (float*)d_out;

  // Workspace: Q,K row-major + V transposed (bf16, 32 MB each), bf16 weights,
  // row norms.
  bf16*  Qb  = (bf16*)d_ws;
  bf16*  Kb  = Qb + (size_t)BN * Dd;
  bf16*  Vtg = Kb + (size_t)BN * Dd;
  bf16*  Wqb = Vtg + (size_t)BN * Dd;
  bf16*  Wkb = Wqb + (size_t)Dd * Dd;
  bf16*  Wvb = Wkb + (size_t)Dd * Dd;
  float* q2  = (float*)(Wvb + (size_t)Dd * Dd);
  float* k2  = q2 + BN;

  wcvt_kernel<<<(Dd * Dd / 8) / 256, 256, 0, stream>>>(Wq, Wk, Wv, Wqb, Wkb, Wvb);
  qkv_kernel<<<BN / 16, 256, 0, stream>>>(x, Wqb, Wkb, Wvb, Qb, Kb, Vtg);
  norms_kernel<<<BN / 8, 256, 0, stream>>>(Qb, Kb, q2, k2);

  const size_t smem =
      (size_t)(16 * Dd + KB * Dd + (size_t)Dd * KB + 16 * KB) * sizeof(bf16) +
      32 * sizeof(float);                   // ~283 KB, fits 320 KB WGP LDS
  (void)hipFuncSetAttribute(reinterpret_cast<const void*>(attn_kernel),
                            hipFuncAttributeMaxDynamicSharedMemorySize, (int)smem);
  attn_kernel<<<BN / 16, 256, smem, stream>>>(Qb, Kb, Vtg, q2, k2, out);
}
